// SelfAttention_6064493822280
// MI455X (gfx1250) — matrix operand order repaired and checked by reading, not compiled, not run on hardware
//
#include <hip/hip_runtime.h>
#include <hip/hip_bf16.h>

// ---------------------------------------------------------------------------
// CDNA5 / gfx1250 fused self-attention:
//   q,k,v = O0 @ {W,U,H}^T + b      (WMMA f16->f32, TDM-staged tiles)
//   O     = softmax(q k^T) v        (flash-attention, TDM double-buffered K)
//   out   = O @ fc0_w^T + fc0_b     (WMMA f16->f32)
// ---------------------------------------------------------------------------

typedef __attribute__((ext_vector_type(16))) _Float16     v16h;
typedef __attribute__((ext_vector_type(8)))  _Float16     v8h;
typedef __attribute__((ext_vector_type(8)))  float        v8f;
typedef __attribute__((ext_vector_type(4)))  unsigned int u32x4;
typedef __attribute__((ext_vector_type(8)))  unsigned int u32x8;

#define THREADS 256        // 8 waves (wave32)
#define BM      128        // M rows per block
#define BN      64         // N cols per block (GEMM kernels)
#define BKG     64         // K per GEMM stage (2 WMMA K-steps)
#define FA_BKV  32         // KV tile in flash attention
#define HIDD    256        // hidden dim

// ---- Tensor Data Mover: 2D tile (f16) global -> LDS -----------------------
// D# layout per CDNA5 ISA. One DMA per call,
// issued once per wave (EXEC ignored); tracked with TENSORcnt.
__device__ __forceinline__ unsigned lds_lo32(const void* p) {
  return (unsigned)(unsigned long long)(uintptr_t)p;  // LDS byte address is the low 32 bits
}

__device__ __forceinline__ void tdm_load_tile_f16(
    const _Float16* gsrc, void* ldst, unsigned tile_d0, unsigned tile_d1,
    unsigned tensor_d0, unsigned tensor_d1, unsigned stride_d0) {
  const unsigned long long ga = (unsigned long long)(uintptr_t)gsrc;
  u32x4 g0;
  g0[0] = 1u;                                           // count=1 (valid D#)
  g0[1] = lds_lo32(ldst);                               // lds_addr (bytes)
  g0[2] = (unsigned)(ga & 0xffffffffu);                 // global_addr[31:0]
  g0[3] = (unsigned)((ga >> 32) & 0x01ffffffu) | (2u << 30);  // [56:32]|type=2
  u32x8 g1;
  g1[0] = 1u << 16;                                     // data_size=1 (2 bytes)
  g1[1] = (tensor_d0 & 0xffffu) << 16;                  // tensor_dim0[15:0]
  g1[2] = (tensor_d0 >> 16) | ((tensor_d1 & 0xffffu) << 16);
  g1[3] = (tensor_d1 >> 16) | (tile_d0 << 16);          // tile_dim0
  g1[4] = tile_d1;                                      // tile_dim1, tile_dim2=0
  g1[5] = stride_d0;                                    // tensor_dim0_stride lo
  g1[6] = 0u;
  g1[7] = 0u;
  asm volatile("tensor_load_to_lds %0, %1" ::"s"(g0), "s"(g1) : "memory");
}

// ---- WMMA fragment loaders (layouts per CDNA5 ISA 7.12.2, wave32) ---------
__device__ __forceinline__ v16h load_fragA(const _Float16* base, int lane, int ld) {
  const int row = lane & 15;
  const int kb  = (lane >> 4) << 3;
  const _Float16* p = base + row * ld + kb;
  const v8h lo = *(const v8h*)p;
  const v8h hi = *(const v8h*)(p + 16);
  v16h r;
#pragma unroll
  for (int i = 0; i < 8; ++i) { r[i] = lo[i]; r[i + 8] = hi[i]; }
  return r;
}

__device__ __forceinline__ v16h load_fragB(const _Float16* base, int lane, int ld) {
  const int col = lane & 15;
  const int kb  = (lane >> 4) << 3;
  const _Float16* p = base + col * ld + kb;
  const v8h lo = *(const v8h*)p;
  const v8h hi = *(const v8h*)(p + 16);
  v16h r;
#pragma unroll
  for (int i = 0; i < 8; ++i) { r[i] = lo[i]; r[i + 8] = hi[i]; }
  return r;
}

__device__ __forceinline__ v8f wmma32(v16h a, v16h b, v8f c) {
  return __builtin_amdgcn_wmma_f32_16x16x32_f16(false, a, false, b,
                                                (short)0, c, false, false);
}

// ---- f32 -> f16 conversion ------------------------------------------------
__global__ void __launch_bounds__(THREADS)
f32_to_f16_kernel(const float* __restrict__ in, _Float16* __restrict__ out, int n) {
  for (int i = blockIdx.x * blockDim.x + threadIdx.x; i < n;
       i += gridDim.x * blockDim.x)
    out[i] = (_Float16)in[i];
}

// ---- GEMM: out[M,Ncols] = X[M,K] @ W[Ncols,K]^T + bias --------------------
// 128x64 tile, 8 waves; TDM double-buffered staging, BK=64 per stage.
__global__ void __launch_bounds__(THREADS)
gemm_xwT_kernel(const _Float16* __restrict__ X, const _Float16* __restrict__ W,
                const float* __restrict__ bias, _Float16* __restrict__ outH,
                float* __restrict__ outF, int Ncols, int K) {
  __shared__ _Float16 sx[2][BM * BKG];   // 2 x 16 KB
  __shared__ _Float16 sw[2][BN * BKG];   // 2 x  8 KB
  const int tid   = threadIdx.x;
  const int wave  = tid >> 5;
  const int lane  = tid & 31;
  const int mbase = blockIdx.x * BM;
  const int nbase = blockIdx.y * BN;

  // prologue: DMA first K-stage into buffer 0
  if (wave == 0) {
    tdm_load_tile_f16(X + (size_t)mbase * K, &sx[0][0], BKG, BM, K, 1u << 20, K);
    tdm_load_tile_f16(W + (size_t)nbase * K, &sw[0][0], BKG, BN, K, 1u << 20, K);
    __builtin_amdgcn_s_wait_tensorcnt(0);
  }
  __syncthreads();

  v8f acc[4] = {};
  for (int k0 = 0; k0 < K; k0 += BKG) {
    const int ib = (k0 / BKG) & 1;
    if ((k0 + BKG < K) && wave == 0) {  // prefetch next stage under compute
      tdm_load_tile_f16(X + (size_t)mbase * K + (k0 + BKG), &sx[ib ^ 1][0],
                        BKG, BM, K, 1u << 20, K);
      tdm_load_tile_f16(W + (size_t)nbase * K + (k0 + BKG), &sw[ib ^ 1][0],
                        BKG, BN, K, 1u << 20, K);
    }
#pragma unroll
    for (int c = 0; c < 2; ++c) {
      const v16h a = load_fragA(&sx[ib][0] + wave * 16 * BKG + c * 32, lane, BKG);
#pragma unroll
      for (int nt = 0; nt < 4; ++nt) {
        const v16h b = load_fragB(&sw[ib][0] + nt * 16 * BKG + c * 32, lane, BKG);
        acc[nt] = wmma32(a, b, acc[nt]);
      }
    }
    if (wave == 0) __builtin_amdgcn_s_wait_tensorcnt(0);  // drain prefetch
    __syncthreads();
  }
  // epilogue: bias + store (C layout: col = lane%16, rows r + 8*(lane/16))
#pragma unroll
  for (int nt = 0; nt < 4; ++nt) {
    const int col = nbase + nt * 16 + (lane & 15);
    const float bv = bias[col];
#pragma unroll
    for (int r = 0; r < 8; ++r) {
      const int row = mbase + wave * 16 + r + ((lane >> 4) << 3);
      const float v = acc[nt][r] + bv;
      if (outH) outH[(size_t)row * Ncols + col] = (_Float16)v;
      else      outF[(size_t)row * Ncols + col] = v;
    }
  }
}

// ---- Flash attention: O = softmax(q k^T) v (no scaling, per reference) ----
// Dynamic LDS layout (halfs): K tiles (TDM, x2), V^T tiles (x2), P staging.
#define FA_TILE (FA_BKV * HIDD)                    // one KV tile, halfs
#define FA_LDS_BYTES ((4 * FA_TILE + 8 * 16 * FA_BKV) * 2)

__device__ __forceinline__ void stage_vt(_Float16* dst, const _Float16* Vh,
                                         int kv, int tid) {
  // svt[d][j] = V[kv+j][d]; coalesced global reads, transposed LDS writes
  for (int idx = tid; idx < FA_BKV * HIDD; idx += THREADS) {
    const int j = idx >> 8;
    const int d = idx & 255;
    dst[d * FA_BKV + j] = Vh[(size_t)(kv + j) * HIDD + d];
  }
}

__global__ void __launch_bounds__(THREADS)
flash_attn_kernel(const _Float16* __restrict__ Q, const _Float16* __restrict__ Kh,
                  const _Float16* __restrict__ Vh, _Float16* __restrict__ Oh,
                  int N) {
  extern __shared__ _Float16 fa_smem[];
  const int tid   = threadIdx.x;
  const int wave  = tid >> 5;
  const int lane  = tid & 31;
  const int qbase = blockIdx.x * BM + wave * 16;
  // runtime pointer arithmetic only (no constant-initialized LDS ptr arrays)
  _Float16* pw = fa_smem + 4 * FA_TILE + wave * (16 * FA_BKV);

  // q rows of this wave, pre-packed as 8 A-fragments (K = 256)
  v16h qf[8];
#pragma unroll
  for (int c = 0; c < 8; ++c)
    qf[c] = load_fragA(Q + (size_t)qbase * HIDD + c * 32, lane, HIDD);

  v8f oacc[16] = {};
  float mrow[8], lrow[8];
#pragma unroll
  for (int r = 0; r < 8; ++r) { mrow[r] = -1e30f; lrow[r] = 0.f; }

  // prologue: tile 0
  if (wave == 0)
    tdm_load_tile_f16(Kh, fa_smem, HIDD, FA_BKV, HIDD, (unsigned)N, HIDD);
  stage_vt(fa_smem + 2 * FA_TILE, Vh, 0, tid);
  if (wave == 0) __builtin_amdgcn_s_wait_tensorcnt(0);
  __syncthreads();

  for (int kv = 0; kv < N; kv += FA_BKV) {
    const int ib = (kv / FA_BKV) & 1;
    _Float16* sk  = fa_smem + ib * FA_TILE;
    _Float16* svt = fa_smem + (2 + ib) * FA_TILE;
    if (kv + FA_BKV < N) {  // prefetch next KV tile under compute
      if (wave == 0)
        tdm_load_tile_f16(Kh + (size_t)(kv + FA_BKV) * HIDD,
                          fa_smem + (ib ^ 1) * FA_TILE,
                          HIDD, FA_BKV, HIDD, (unsigned)N, HIDD);
      stage_vt(fa_smem + (2 + (ib ^ 1)) * FA_TILE, Vh, kv + FA_BKV, tid);
    }

    // S = q k^T : 16x32 (two 16x16 C tiles), K loop over 256
    v8f s0 = {}, s1 = {};
#pragma unroll
    for (int c = 0; c < 8; ++c) {
      const v16h b0 = load_fragB(sk + c * 32, lane, HIDD);
      const v16h b1 = load_fragB(sk + 16 * HIDD + c * 32, lane, HIDD);
      s0 = wmma32(qf[c], b0, s0);
      s1 = wmma32(qf[c], b1, s1);
    }

    // online softmax (row reductions stay within each 16-lane half)
    float scale[8];
#pragma unroll
    for (int r = 0; r < 8; ++r) {
      float mx = fmaxf(s0[r], s1[r]);
      mx = fmaxf(mx, __shfl_xor(mx, 1));
      mx = fmaxf(mx, __shfl_xor(mx, 2));
      mx = fmaxf(mx, __shfl_xor(mx, 4));
      mx = fmaxf(mx, __shfl_xor(mx, 8));
      const float mnew = fmaxf(mrow[r], mx);
      const float sc = __expf(mrow[r] - mnew);
      const float p0 = __expf(s0[r] - mnew);
      const float p1 = __expf(s1[r] - mnew);
      float rs = p0 + p1;
      rs += __shfl_xor(rs, 1);
      rs += __shfl_xor(rs, 2);
      rs += __shfl_xor(rs, 4);
      rs += __shfl_xor(rs, 8);
      lrow[r] = lrow[r] * sc + rs;
      mrow[r] = mnew;
      scale[r] = sc;
      s0[r] = p0;
      s1[r] = p1;
    }
#pragma unroll
    for (int nt = 0; nt < 16; ++nt)
#pragma unroll
      for (int r = 0; r < 8; ++r) oacc[nt][r] *= scale[r];

    // C-layout -> A-layout for P via per-wave LDS staging
#pragma unroll
    for (int r = 0; r < 8; ++r) {
      const int row = r + ((lane >> 4) << 3);
      pw[row * FA_BKV + (lane & 15)]      = (_Float16)s0[r];
      pw[row * FA_BKV + 16 + (lane & 15)] = (_Float16)s1[r];
    }
    asm volatile("s_wait_dscnt 0" ::: "memory");  // same-wave LDS RAW

    // O += P @ V : one K=32 step, 16 N subtiles over HID=256
    const v16h pa = load_fragA(pw, lane, FA_BKV);
#pragma unroll
    for (int nt = 0; nt < 16; ++nt) {
      const v16h b = load_fragB(svt + nt * 16 * FA_BKV, lane, FA_BKV);
      oacc[nt] = wmma32(pa, b, oacc[nt]);
    }
    if (wave == 0) __builtin_amdgcn_s_wait_tensorcnt(0);  // drain prefetch DMA
    __syncthreads();
  }

  // normalize and emit f16 attention output
  float inv[8];
#pragma unroll
  for (int r = 0; r < 8; ++r) inv[r] = 1.0f / lrow[r];
#pragma unroll
  for (int nt = 0; nt < 16; ++nt) {
    const int col = nt * 16 + (lane & 15);
#pragma unroll
    for (int r = 0; r < 8; ++r) {
      const int row = qbase + r + ((lane >> 4) << 3);
      Oh[(size_t)row * HIDD + col] = (_Float16)(oacc[nt][r] * inv[r]);
    }
  }
}

// ---------------------------------------------------------------------------
extern "C" void kernel_launch(void* const* d_in, const int* in_sizes, int n_in,
                              void* d_out, int out_size, void* d_ws, size_t ws_size,
                              hipStream_t stream) {
  (void)in_sizes; (void)n_in; (void)out_size; (void)ws_size;
  const int N = 8192, IN = 512, HID = 256, OUT = 512;

  const float* O0    = (const float*)d_in[0];
  const float* W_w   = (const float*)d_in[1];
  const float* W_b   = (const float*)d_in[2];
  const float* U_w   = (const float*)d_in[3];
  const float* U_b   = (const float*)d_in[4];
  const float* H_w   = (const float*)d_in[5];
  const float* H_b   = (const float*)d_in[6];
  const float* fc0_w = (const float*)d_in[7];
  const float* fc0_b = (const float*)d_in[8];

  char* ws = (char*)d_ws;
  size_t off = 0;
  auto take = [&](size_t nhalf) -> _Float16* {
    _Float16* p = (_Float16*)(ws + off);
    off += (nhalf * sizeof(_Float16) + 255) & ~(size_t)255;
    return p;
  };
  _Float16* O0h = take((size_t)N * IN);
  _Float16* Wh  = take((size_t)HID * IN);
  _Float16* Uh  = take((size_t)HID * IN);
  _Float16* Hh  = take((size_t)HID * IN);
  _Float16* Fh  = take((size_t)OUT * HID);
  _Float16* qh  = take((size_t)N * HID);
  _Float16* kh  = take((size_t)N * HID);
  _Float16* vh  = take((size_t)N * HID);
  _Float16* ah  = take((size_t)N * HID);

  auto cvt = [&](const float* src, _Float16* dst, int n) {
    f32_to_f16_kernel<<<dim3((n + THREADS - 1) / THREADS), dim3(THREADS), 0,
                        stream>>>(src, dst, n);
  };
  cvt(O0, O0h, N * IN);
  cvt(W_w, Wh, HID * IN);
  cvt(U_w, Uh, HID * IN);
  cvt(H_w, Hh, HID * IN);
  cvt(fc0_w, Fh, OUT * HID);

  const dim3 blk(THREADS);
  const dim3 gproj(N / BM, HID / BN);
  gemm_xwT_kernel<<<gproj, blk, 0, stream>>>(O0h, Wh, W_b, qh, nullptr, HID, IN);
  gemm_xwT_kernel<<<gproj, blk, 0, stream>>>(O0h, Uh, U_b, kh, nullptr, HID, IN);
  gemm_xwT_kernel<<<gproj, blk, 0, stream>>>(O0h, Hh, H_b, vh, nullptr, HID, IN);

  flash_attn_kernel<<<dim3(N / BM), blk, FA_LDS_BYTES, stream>>>(qh, kh, vh, ah, N);

  const dim3 gfc(N / BM, OUT / BN);
  gemm_xwT_kernel<<<gfc, blk, 0, stream>>>(ah, Fh, fc0_b, nullptr, (float*)d_out,
                                           OUT, HID);
}
